// GARNN_42743514529905
// MI455X (gfx1250) — matrix operand order repaired and checked by reading, not compiled, not run on hardware
//
#include <hip/hip_runtime.h>

typedef unsigned short u16;
typedef __attribute__((ext_vector_type(16))) __bf16 v16bf;
typedef __attribute__((ext_vector_type(2)))  __bf16 v2bf;
typedef __attribute__((ext_vector_type(8)))  float  v8f;

namespace {

constexpr int cB = 16, cT = 12, cN = 325, cD = 64, cL = 2, cG = 192;
constexpr int NPAD = 336;     // 21 tiles of 16 rows
constexpr int NT   = 21;      // row/col tiles over N
constexpr int GTI  = 12;      // 16-col tiles over G=192
constexpr int VST  = 336;     // vT row stride (elements)
constexpr int THREADS = 672;  // 21 waves, one wave per row tile

// score scale: (1/sqrt(64)) * log2(e)  -> exp2 domain softmax
constexpr float SCL = 0.125f * 1.44269504088896340736f;

// LDS layout (bytes, all 16B aligned)
constexpr int XS_OFF  = 0;                          // u16[336*64]
constexpr int KB_OFF  = XS_OFF + NPAD * cD * 2;     // u16[336*64]
constexpr int VT_OFF  = KB_OFF + NPAD * cD * 2;     // u16[192*336]  (v transposed)
constexpr int WQ_OFF  = VT_OFF + cG * VST * 2;      // u16[64*64]    (Wq transposed)
constexpr int WK_OFF  = WQ_OFF + cD * cD * 2;       // u16[64*64]
constexpr int WV_OFF  = WK_OFF + cD * cD * 2;       // u16[192*64]
constexpr int BS_OFF  = WV_OFF + cG * cD * 2;       // float[192]
constexpr int LG_OFF  = BS_OFF + cG * 4;            // float[2*64]
constexpr int LB_OFF  = LG_OFF + cL * cD * 4;       // float[2*64]
constexpr int STG_OFF = LB_OFF + cL * cD * 4;       // u16[21][1024] per-wave staging
constexpr int LDS_BYTES = STG_OFF + NT * 1024 * 2;  // 300800 B < 320 KB

struct alignas(16) U4 { unsigned int x, y, z, w; };
union AB32 { U4 q[2]; v16bf v; };

__device__ __forceinline__ u16 f2bf(float f) {
  __bf16 h = (__bf16)f;                 // native f32->bf16 convert (RNE)
  return __builtin_bit_cast(u16, h);
}

// two f32 -> packed 2x bf16 in one dword (one v_cvt_pk_bf16_f32 if available)
__device__ __forceinline__ unsigned pk_bf16(float a, float b) {
#if __has_builtin(__builtin_amdgcn_cvt_pk_bf16_f32)
  v2bf r = __builtin_amdgcn_cvt_pk_bf16_f32(a, b);
  return __builtin_bit_cast(unsigned, r);
#else
  return (unsigned)f2bf(a) | ((unsigned)f2bf(b) << 16);
#endif
}

__device__ __forceinline__ v8f wmma_bf16(v16bf a, v16bf b, v8f c) {
  return __builtin_amdgcn_wmma_f32_16x16x32_bf16(false, a, false, b, (short)0, c,
                                                 false, false);
}

// A-fragment (16x32 bf16) from row-major LDS: row m = lane&15 (passed as global row),
// element e -> K = k0 + e + 16*half   (contiguous 32B)
__device__ __forceinline__ v16bf ldsA(const u16* base, int row, int stride, int k0,
                                      int half) {
  AB32 t;
  const u16* p = base + row * stride + k0 + 16 * half;
  t.q[0] = *reinterpret_cast<const U4*>(p);
  t.q[1] = *reinterpret_cast<const U4*>(p + 8);
  return t.v;
}

// B-fragment (32x16 bf16): element e -> K = e + 16*half, contiguous 32B at p
__device__ __forceinline__ v16bf ldsB(const u16* p) {
  AB32 t;
  t.q[0] = *reinterpret_cast<const U4*>(p);
  t.q[1] = *reinterpret_cast<const U4*>(p + 8);
  return t.v;
}

__device__ __forceinline__ float redmax16(float v) {
#pragma unroll
  for (int m = 1; m < 16; m <<= 1) v = fmaxf(v, __shfl_xor(v, m, 32));
  return v;
}
__device__ __forceinline__ float redsum16(float v) {
#pragma unroll
  for (int m = 1; m < 16; m <<= 1) v += __shfl_xor(v, m, 32);
  return v;
}

// One full graph-attention pass: out = softmax(xWq (xWk)^T / 8) (xWv) + b
// x is in LDS (bf16). Each wave owns row tile `wid`; O holds its 16xG f32 result.
__device__ __forceinline__ void gat_compute(
    const float* __restrict__ Wq, const float* __restrict__ Wk,
    const float* __restrict__ Wv, const float* __restrict__ bb,
    u16* xs, u16* kb, u16* vt, u16* wqT, u16* wkT, u16* wvT, float* bsh,
    u16* mystg, float* __restrict__ attnB,
    int tid, int wid, int nn, int half, v8f (&O)[GTI]) {
  // ---- stage transposed weights + bias ----
  for (int idx = tid; idx < cD * cD; idx += THREADS) {
    int o = idx >> 6, i = idx & 63;
    wqT[idx] = f2bf(Wq[i * cD + o]);
    wkT[idx] = f2bf(Wk[i * cD + o]);
  }
  for (int idx = tid; idx < cG * cD; idx += THREADS) {
    int o = idx >> 6, i = idx & 63;
    wvT[idx] = f2bf(Wv[i * cG + o]);
  }
  if (tid < cG) bsh[tid] = bb[tid];
  __syncthreads();

  // ---- k = x@Wk (row-major bf16 in LDS) ----
  for (int tl = wid; tl < NT * 4; tl += NT) {
    int it = tl >> 2, ct = tl & 3;
    v8f c = {};
    c = wmma_bf16(ldsA(xs, it * 16 + nn, cD, 0, half),
                  ldsB(wkT + (ct * 16 + nn) * cD + 0 + 16 * half), c);
    c = wmma_bf16(ldsA(xs, it * 16 + nn, cD, 32, half),
                  ldsB(wkT + (ct * 16 + nn) * cD + 32 + 16 * half), c);
#pragma unroll
    for (int r = 0; r < 8; r += 2) {
      unsigned pq = pk_bf16(c[r], c[r + 1]);
      u16* p0 = &kb[(it * 16 + r + 8 * half) * cD + ct * 16 + nn];
      p0[0]  = (u16)pq;          // row r
      p0[cD] = (u16)(pq >> 16);  // row r+1
    }
  }
  // ---- vT = (x@Wv)^T (bf16, packed 16B stores) ----
  for (int tl = wid; tl < NT * GTI; tl += NT) {
    int it = tl / GTI, gt = tl % GTI;
    v8f c = {};
    c = wmma_bf16(ldsA(xs, it * 16 + nn, cD, 0, half),
                  ldsB(wvT + (gt * 16 + nn) * cD + 0 + 16 * half), c);
    c = wmma_bf16(ldsA(xs, it * 16 + nn, cD, 32, half),
                  ldsB(wvT + (gt * 16 + nn) * cD + 32 + 16 * half), c);
    U4 pk;
    pk.x = pk_bf16(c[0], c[1]);
    pk.y = pk_bf16(c[2], c[3]);
    pk.z = pk_bf16(c[4], c[5]);
    pk.w = pk_bf16(c[6], c[7]);
    *reinterpret_cast<U4*>(vt + (gt * 16 + nn) * VST + it * 16 + 8 * half) = pk;
  }
  // ---- q tile for this wave: compute, transpose via staging, keep A-frags ----
#pragma unroll
  for (int ct = 0; ct < 4; ++ct) {
    v8f c = {};
    c = wmma_bf16(ldsA(xs, wid * 16 + nn, cD, 0, half),
                  ldsB(wqT + (ct * 16 + nn) * cD + 0 + 16 * half), c);
    c = wmma_bf16(ldsA(xs, wid * 16 + nn, cD, 32, half),
                  ldsB(wqT + (ct * 16 + nn) * cD + 32 + 16 * half), c);
#pragma unroll
    for (int r = 0; r < 8; r += 2) {
      unsigned pq = pk_bf16(c[r], c[r + 1]);
      u16* p0 = &mystg[(r + 8 * half) * cD + ct * 16 + nn];
      p0[0]  = (u16)pq;
      p0[cD] = (u16)(pq >> 16);
    }
  }
  v16bf qf0 = ldsA(mystg, nn, cD, 0, half);
  v16bf qf1 = ldsA(mystg, nn, cD, 32, half);
  __syncthreads();

  // ---- pass 1: per-lane online softmax stats (no cross-lane ops in loop) ----
  float ml[8], ll[8];
#pragma unroll
  for (int r = 0; r < 8; ++r) { ml[r] = -3.0e38f; ll[r] = 0.f; }
  for (int j = 0; j < NT; ++j) {
    v8f s = {};
    s = wmma_bf16(qf0, ldsB(kb + (j * 16 + nn) * cD + 0 + 16 * half), s);
    s = wmma_bf16(qf1, ldsB(kb + (j * 16 + nn) * cD + 32 + 16 * half), s);
    bool cv = (j * 16 + nn) < cN;
#pragma unroll
    for (int r = 0; r < 8; ++r) {
      float sv = cv ? s[r] * SCL : -3.0e38f;
      float nm = fmaxf(ml[r], sv);
      float p = cv ? exp2f(sv - nm) : 0.f;
      ll[r] = ll[r] * exp2f(ml[r] - nm) + p;
      ml[r] = nm;
    }
  }
  // one cross-lane reduction per row; reciprocal once (kills pass-2 divides)
  float mrow[8], linv[8];
#pragma unroll
  for (int r = 0; r < 8; ++r) {
    float M = redmax16(ml[r]);
    float L = redsum16(ll[r] * exp2f(ml[r] - M));
    mrow[r] = M;
    linv[r] = 1.0f / L;
  }

  // ---- pass 2: P = exp2(s'-m)*linv, emit attn (NT stores), O = P@v + b ----
#pragma unroll
  for (int g = 0; g < GTI; ++g) {
    float bv = bsh[g * 16 + nn];
    v8f c;
#pragma unroll
    for (int r = 0; r < 8; ++r) c[r] = bv;
    O[g] = c;
  }
  for (int jp = 0; jp < 11; ++jp) {
#pragma unroll
    for (int jj = 0; jj < 2; ++jj) {
      int j = jp * 2 + jj;
      if (j < NT) {
        v8f s = {};
        s = wmma_bf16(qf0, ldsB(kb + (j * 16 + nn) * cD + 0 + 16 * half), s);
        s = wmma_bf16(qf1, ldsB(kb + (j * 16 + nn) * cD + 32 + 16 * half), s);
        int col = j * 16 + nn;
        bool cv = col < cN;
#pragma unroll
        for (int r = 0; r < 8; r += 2) {
          float p0 = cv ? exp2f(s[r] * SCL - mrow[r]) * linv[r] : 0.f;
          float p1 = cv ? exp2f(s[r + 1] * SCL - mrow[r + 1]) * linv[r + 1] : 0.f;
          int row0 = wid * 16 + r + 8 * half;
          if (cv && row0 < cN)
            __builtin_nontemporal_store(p0, &attnB[(size_t)row0 * cN + col]);
          if (cv && row0 + 1 < cN)
            __builtin_nontemporal_store(p1, &attnB[(size_t)(row0 + 1) * cN + col]);
          unsigned pq = pk_bf16(p0, p1);
          u16* sp = &mystg[(r + 8 * half) * 32 + jj * 16 + nn];
          sp[0]  = (u16)pq;
          sp[32] = (u16)(pq >> 16);
        }
      } else {
#pragma unroll
        for (int r = 0; r < 8; ++r)
          mystg[(r + 8 * half) * 32 + jj * 16 + nn] = 0;
      }
    }
    v16bf pa = ldsA(mystg, nn, 32, 0, half);
#pragma unroll
    for (int g = 0; g < GTI; ++g)
      O[g] = wmma_bf16(pa, ldsB(vt + (g * 16 + nn) * VST + jp * 32 + 16 * half),
                       O[g]);
  }
  __syncthreads();
}

__global__ void __launch_bounds__(THREADS, 1) garnn_kernel(
    const float* __restrict__ x,
    const float* __restrict__ Wq_i, const float* __restrict__ Wk_i,
    const float* __restrict__ Wv_i, const float* __restrict__ b_i,
    const float* __restrict__ Wq_h, const float* __restrict__ Wk_h,
    const float* __restrict__ Wv_h, const float* __restrict__ b_h,
    const float* __restrict__ ln_g, const float* __restrict__ ln_b,
    float* __restrict__ out_out, float* __restrict__ out_hid,
    float* __restrict__ attn_i, float* __restrict__ attn_h) {
  extern __shared__ char smem[];
  u16* xs   = (u16*)(smem + XS_OFF);
  u16* kb   = (u16*)(smem + KB_OFF);
  u16* vt   = (u16*)(smem + VT_OFF);
  u16* wqT  = (u16*)(smem + WQ_OFF);
  u16* wkT  = (u16*)(smem + WK_OFF);
  u16* wvT  = (u16*)(smem + WV_OFF);
  float* bsh = (float*)(smem + BS_OFF);
  float* lg  = (float*)(smem + LG_OFF);
  float* lb  = (float*)(smem + LB_OFF);
  u16* stg  = (u16*)(smem + STG_OFF);

  const int tid = threadIdx.x;
  const int wid = tid >> 5;
  const int lane = tid & 31;
  const int half = lane >> 4;
  const int nn = lane & 15;
  const int b = blockIdx.x;
  u16* mystg = stg + wid * 1024;

  if (tid < cL * cD) { lg[tid] = ln_g[tid]; lb[tid] = ln_b[tid]; }
  __syncthreads();

  for (int t = 0; t < cT; ++t) {
    // layer-0 input: x[b][t] -> LDS bf16 (pad rows zeroed); streamed once -> NT
    const float* xt = x + ((size_t)(b * cT + t)) * cN * cD;
    for (int idx = tid; idx < NPAD * cD; idx += THREADS) {
      int row = idx >> 6, dcol = idx & 63;
      xs[idx] = (row < cN)
                    ? f2bf(__builtin_nontemporal_load(&xt[row * cD + dcol]))
                    : (u16)0;
    }
    if (t + 1 < cT && tid * 32 < cN * cD)
      __builtin_prefetch(xt + cN * cD + tid * 32, 0, 1);  // global_prefetch_b8
    __syncthreads();

    for (int l = 0; l < cL; ++l) {
      v8f Oi[GTI], Oh[GTI];
      size_t abase = (((size_t)(b * cT + t)) * cL + l) * (size_t)cN * cN;
      gat_compute(Wq_i + l * cD * cD, Wk_i + l * cD * cD, Wv_i + l * cD * cG,
                  b_i + l * cG, xs, kb, vt, wqT, wkT, wvT, bsh, mystg,
                  attn_i + abase, tid, wid, nn, half, Oi);
      gat_compute(Wq_h + l * cD * cD, Wk_h + l * cD * cD, Wv_h + l * cD * cG,
                  b_h + l * cG, xs, kb, vt, wqT, wkT, wvT, bsh, mystg,
                  attn_h + abase, tid, wid, nn, half, Oh);

      // ---- GRU combine + LayerNorm (wave owns rows of tile `wid`) ----
      float* hid = out_hid + ((size_t)(b * cL + l)) * cN * cD;
      float* op  = out_out + ((size_t)(b * cT + t)) * cN * cD;
      float ov[4][8];
#pragma unroll
      for (int c4 = 0; c4 < 4; ++c4) {
#pragma unroll
        for (int r = 0; r < 8; ++r) {
          int row = wid * 16 + r + 8 * half;
          int d = c4 * 16 + nn;
          float h = (row < cN) ? hid[row * cD + d] : 0.f;
          float rg = 1.f / (1.f + __expf(-(Oi[c4][r] + Oh[c4][r])));
          float ig = 1.f / (1.f + __expf(-(Oi[c4 + 4][r] + Oh[c4 + 4][r])));
          float ng = tanhf(Oi[c4 + 8][r] + rg * Oh[c4 + 8][r]);
          ov[c4][r] = ng + ig * (h - ng);
        }
      }
#pragma unroll
      for (int r = 0; r < 8; ++r) {
        float sum = ov[0][r] + ov[1][r] + ov[2][r] + ov[3][r];
        sum = redsum16(sum);
        float mu = sum * (1.f / cD);
        float vs = 0.f;
#pragma unroll
        for (int c4 = 0; c4 < 4; ++c4) { float dd = ov[c4][r] - mu; vs += dd * dd; }
        vs = redsum16(vs);
        float inv = rsqrtf(vs * (1.f / cD) + 1e-5f);
        int row = wid * 16 + r + 8 * half;
#pragma unroll
        for (int c4 = 0; c4 < 4; ++c4) {
          int d = c4 * 16 + nn;
          float on = (ov[c4][r] - mu) * inv * lg[l * cD + d] + lb[l * cD + d];
          if (row < cN) {
            hid[row * cD + d] = on;                                  // re-read: RT
            if (l == cL - 1)
              __builtin_nontemporal_store(on, &op[row * cD + d]);    // write-once
          }
          xs[row * cD + d] = (row < cN) ? f2bf(on) : (u16)0;  // next layer/step in
        }
      }
      __syncthreads();
    }
  }
}

__global__ void zero_f32(float* __restrict__ p, int n) {
  int i = blockIdx.x * blockDim.x + threadIdx.x;
  if (i < n) p[i] = 0.f;
}

}  // namespace

extern "C" void kernel_launch(void* const* d_in, const int* in_sizes, int n_in,
                              void* d_out, int out_size, void* d_ws, size_t ws_size,
                              hipStream_t stream) {
  (void)in_sizes; (void)n_in; (void)out_size; (void)d_ws; (void)ws_size;
  const float* x    = (const float*)d_in[0];
  const float* Wq_i = (const float*)d_in[1];
  const float* Wk_i = (const float*)d_in[2];
  const float* Wv_i = (const float*)d_in[3];
  const float* b_i  = (const float*)d_in[4];
  const float* Wq_h = (const float*)d_in[5];
  const float* Wk_h = (const float*)d_in[6];
  const float* Wv_h = (const float*)d_in[7];
  const float* b_h  = (const float*)d_in[8];
  const float* ln_g = (const float*)d_in[9];
  const float* ln_b = (const float*)d_in[10];

  float* out     = (float*)d_out;
  float* out_out = out;                                         // (B,T,N,D)
  float* out_hid = out_out + (size_t)cB * cT * cN * cD;         // (B,L,N,D)
  float* attn_i  = out_hid + (size_t)cB * cL * cN * cD;         // (B,T,L,N,N)
  float* attn_h  = attn_i + (size_t)cB * cT * (size_t)cL * cN * cN;

  // hidden region doubles as live GRU state; must start at zero every call
  int hn = cB * cL * cN * cD;
  zero_f32<<<(hn + 255) / 256, 256, 0, stream>>>(out_hid, hn);

  (void)hipFuncSetAttribute((const void*)garnn_kernel,
                            hipFuncAttributeMaxDynamicSharedMemorySize, LDS_BYTES);
  garnn_kernel<<<cB, THREADS, LDS_BYTES, stream>>>(
      x, Wq_i, Wk_i, Wv_i, b_i, Wq_h, Wk_h, Wv_h, b_h, ln_g, ln_b,
      out_out, out_hid, attn_i, attn_h);
}
